// MambaBlock_309237646081
// MI455X (gfx1250) — matrix operand order repaired and checked by reading, not compiled, not run on hardware
//
#include <hip/hip_runtime.h>

// Problem constants (from reference)
#define S_LEN 1024
#define BSZ   16
#define DMDL  1024
#define DINN  2048   // DI = DM*EXPAND
#define DSTT  16     // DS
#define DTRK  64     // DTR
#define NXP   96     // DTR + 2*DS

typedef __attribute__((ext_vector_type(16))) __bf16 v16bf;
typedef __attribute__((ext_vector_type(8)))  float  v8f;
typedef __attribute__((ext_vector_type(4)))  unsigned int u32x4;

union FragB16 { v16bf v; u32x4 q[2]; unsigned int u[8]; unsigned short h[16]; };
union AccF    { v8f v; float f[8]; };

__device__ __forceinline__ unsigned short f2b(float x) {
  unsigned u = __float_as_uint(x);
  unsigned r = (u >> 16) & 1u;              // round-to-nearest-even
  return (unsigned short)((u + 0x7fffu + r) >> 16);
}
__device__ __forceinline__ float b2f(unsigned short h) {
  return __uint_as_float(((unsigned)h) << 16);
}

// ---------------------------------------------------------------------------
// Pack a KxN row-major f32 weight into WMMA B-fragment tiles (bf16).
// Tile (kb,nb) = 32 lanes x 16 bf16: lane l -> n = nb*16+(l&15),
// k = kb*32 + (e>>3)*16 + (l>>4)*8 + (e&7)  (ISA 7.12.2 16-bit B layout).
// ---------------------------------------------------------------------------
__global__ __launch_bounds__(256) void pack_w(const float* __restrict__ src,
                                              unsigned short* __restrict__ dst,
                                              int K, int N) {
  size_t total = (size_t)K * N;
  int ntn = N >> 4;
  for (size_t idx = (size_t)blockIdx.x * blockDim.x + threadIdx.x; idx < total;
       idx += (size_t)gridDim.x * blockDim.x) {
    int e    = (int)(idx & 15);
    int lane = (int)((idx >> 4) & 31);
    size_t tile = idx >> 9;
    int nb = (int)(tile % ntn);
    int kb = (int)(tile / ntn);
    int n = nb * 16 + (lane & 15);
    int k = kb * 32 + ((e >> 3) << 4) + ((lane >> 4) << 3) + (e & 7);
    dst[idx] = f2b(src[(size_t)k * N + n]);
  }
}

__global__ __launch_bounds__(256) void cvt_bf16(const float* __restrict__ src,
                                                unsigned short* __restrict__ dst,
                                                size_t n) {
  for (size_t i = (size_t)blockIdx.x * blockDim.x + threadIdx.x; i < n;
       i += (size_t)gridDim.x * blockDim.x)
    dst[i] = f2b(src[i]);
}

// A-fragment (16x32 bf16) gather from row-major bf16, per ISA 7.12.2.
// Contiguous dword pattern -> compiler coalesces into 2x b128 per lane.
__device__ __forceinline__ void load_a(FragB16& a, const unsigned short* __restrict__ base,
                                       int ld, int m, int kh) {
#pragma unroll
  for (int p = 0; p < 8; ++p) {
    int k = ((p & 3) << 1) + ((p >> 2) << 4) + (kh << 3);
    a.u[p] = *(const unsigned int*)(base + (size_t)m * ld + k);
  }
}
__device__ __forceinline__ void load_b(FragB16& b, const unsigned short* __restrict__ tb) {
  b.q[0] = *(const u32x4*)(tb);
  b.q[1] = *(const u32x4*)(tb + 8);
}

// ---------------------------------------------------------------------------
// LDS-staged 16x128 GEMM core. Block = 8 waves sharing one column group.
// Per K-step: wave wv fetches B tile (kb, nb*8+wv) global->regs->LDS (double
// buffered, 2x8KB), barrier, then all 8 waves ds_load the 8 fragments and run
// 8 WMMAs while the NEXT K-step's global fetch is already in flight.
// A fragments: per-wave global, ping-pong a0/a1 (k-unrolled by 2, no copies).
// ---------------------------------------------------------------------------
template <int TNB>
__device__ __forceinline__ void gemm_core_lds(const unsigned short* __restrict__ arow,
                                              int lda,
                                              const unsigned short* __restrict__ btiles,
                                              unsigned short* __restrict__ sB,  // [2*8*512]
                                              AccF (&acc)[8], int K32,
                                              int m, int kh, int lane, int wv) {
  const unsigned short* gsrc = btiles + (size_t)wv * 512 + lane * 16;  // (kb=0, j=wv)
  u32x4 s0 = *(const u32x4*)(gsrc);
  u32x4 s1 = *(const u32x4*)(gsrc + 8);
  FragB16 a0, a1;
  load_a(a0, arow, lda, m, kh);

  for (int kb = 0; kb < K32; kb += 2) {   // K32 even
    { // ---- step kb: buffer 0, consumes a0, prefetches a1 ----
      unsigned short* sdst = sB + (size_t)wv * 512 + lane * 16;
      *(u32x4*)(sdst) = s0;
      *(u32x4*)(sdst + 8) = s1;
      __syncthreads();
      const int kb1 = kb + 1;
      const unsigned short* g = btiles + ((size_t)kb1 * TNB + wv) * 512 + lane * 16;
      s0 = *(const u32x4*)(g);
      s1 = *(const u32x4*)(g + 8);
      load_a(a1, arow + kb1 * 32, lda, m, kh);
      const unsigned short* sbuf = sB + lane * 16;
#pragma unroll
      for (int j = 0; j < 8; ++j) {
        FragB16 b;
        b.q[0] = *(const u32x4*)(sbuf + j * 512);
        b.q[1] = *(const u32x4*)(sbuf + j * 512 + 8);
        acc[j].v = __builtin_amdgcn_wmma_f32_16x16x32_bf16(false, a0.v, false, b.v,
                                                           (short)0, acc[j].v, false, false);
      }
    }
    { // ---- step kb+1: buffer 1, consumes a1, prefetches a0 ----
      unsigned short* sdst = sB + (size_t)(8 + wv) * 512 + lane * 16;
      *(u32x4*)(sdst) = s0;
      *(u32x4*)(sdst + 8) = s1;
      __syncthreads();
      const int kb2 = (kb + 2 < K32) ? kb + 2 : 0;   // clamped wrap, no branch
      const unsigned short* g = btiles + ((size_t)kb2 * TNB + wv) * 512 + lane * 16;
      s0 = *(const u32x4*)(g);
      s1 = *(const u32x4*)(g + 8);
      load_a(a0, arow + kb2 * 32, lda, m, kh);
      const unsigned short* sbuf = sB + (size_t)8 * 512 + lane * 16;
#pragma unroll
      for (int j = 0; j < 8; ++j) {
        FragB16 b;
        b.q[0] = *(const u32x4*)(sbuf + j * 512);
        b.q[1] = *(const u32x4*)(sbuf + j * 512 + 8);
        acc[j].v = __builtin_amdgcn_wmma_f32_16x16x32_bf16(false, a1.v, false, b.v,
                                                           (short)0, acc[j].v, false, false);
      }
    }
  }
}

// ---------------------------------------------------------------------------
// K2: xz = x @ W_in  (M=16384, K=1024, N=4096). One wave -> 16x128 tile.
// 8 waves/block share one column group (B staged once via LDS), 8
// consecutive t. Cols [0,2048)->x_path, [2048,4096)->z (bf16).
// ---------------------------------------------------------------------------
__global__ __launch_bounds__(256) void gemm_xz(const unsigned short* __restrict__ xh,
                                               const unsigned short* __restrict__ Wp,
                                               unsigned short* __restrict__ xpath,
                                               unsigned short* __restrict__ zh) {
  __shared__ unsigned short sB[2 * 8 * 512];   // 16 KB double buffer
  const int lane = threadIdx.x & 31;
  const int wv   = threadIdx.x >> 5;
  const int wid  = (blockIdx.x << 3) + wv;
  const int nb = wid >> 10;    // 32 groups of 128 cols (uniform within block)
  const int t  = wid & 1023;   // 1024 row tiles (one per timestep)
  const int m = lane & 15, kh = lane >> 4;

  AccF acc[8];
#pragma unroll
  for (int j = 0; j < 8; ++j)
#pragma unroll
    for (int r = 0; r < 8; ++r) acc[j].f[r] = 0.f;

  const unsigned short* arow   = xh + (size_t)(t * 16) * DMDL;
  const unsigned short* btiles = Wp + (size_t)(nb * 8) * 512;
  gemm_core_lds<256>(arow, DMDL, btiles, sB, acc, DMDL / 32, m, kh, lane, wv);

#pragma unroll
  for (int j = 0; j < 8; ++j) {
    int col = nb * 128 + j * 16 + (lane & 15);
    unsigned short* dst = (col < DINN) ? xpath : zh;
    int c = (col < DINN) ? col : col - DINN;
#pragma unroll
    for (int r = 0; r < 8; ++r) {
      int row = t * 16 + (kh << 3) + r;
      dst[(size_t)row * DINN + c] = f2b(acc[j].f[r]);
    }
  }
}

// ---------------------------------------------------------------------------
// K3: causal 4-tap conv with done-reset masking + SiLU (fully parallel).
// x_path[t-k] contributes iff dones[t-k .. t-1] are all zero.
// ---------------------------------------------------------------------------
__global__ __launch_bounds__(256) void conv_silu(const unsigned short* __restrict__ xpath,
                                                 const float* __restrict__ conv_w,
                                                 const float* __restrict__ conv_b,
                                                 const int* __restrict__ dones,
                                                 unsigned short* __restrict__ xch) {
  const size_t total = (size_t)S_LEN * BSZ * DINN;
  const size_t rowstride = (size_t)BSZ * DINN;
  for (size_t idx = (size_t)blockIdx.x * blockDim.x + threadIdx.x; idx < total;
       idx += (size_t)gridDim.x * blockDim.x) {
    int di = (int)(idx % DINN);
    size_t rb = idx / DINN;     // t*B + b
    int b = (int)(rb % BSZ);
    int t = (int)(rb / BSZ);
    float w0 = conv_w[di * 4 + 0], w1 = conv_w[di * 4 + 1];
    float w2 = conv_w[di * 4 + 2], w3 = conv_w[di * 4 + 3];
    float acc = conv_b[di] + w3 * b2f(xpath[idx]);
    bool live = true;
    if (t >= 1) {
      live = (dones[(t - 1) * BSZ + b] == 0);
      if (live) acc += w2 * b2f(xpath[idx - rowstride]);
    }
    if (t >= 2 && live) {
      live = (dones[(t - 2) * BSZ + b] == 0);
      if (live) acc += w1 * b2f(xpath[idx - 2 * rowstride]);
    }
    if (t >= 3 && live) {
      live = (dones[(t - 3) * BSZ + b] == 0);
      if (live) acc += w0 * b2f(xpath[idx - 3 * rowstride]);
    }
    float sc = acc / (1.f + __expf(-acc));   // SiLU
    xch[idx] = f2b(sc);
  }
}

// ---------------------------------------------------------------------------
// K4: xp = x_conv @ W_x  (M=16384, K=2048, N=96). One wave -> 16x16 tile,
// ping-pong A/B across the K loop; block shares nb.
// ---------------------------------------------------------------------------
__global__ __launch_bounds__(256) void gemm_xp(const unsigned short* __restrict__ xch,
                                               const unsigned short* __restrict__ Wxp,
                                               float* __restrict__ xp) {
  const int lane = threadIdx.x & 31;
  const int wid  = (blockIdx.x << 3) + (threadIdx.x >> 5);
  const int nb = wid >> 10;    // 6 col tiles (shared within block)
  const int t  = wid & 1023;
  const int m = lane & 15, kh = lane >> 4;
  const int K32 = DINN / 32;

  AccF acc;
#pragma unroll
  for (int r = 0; r < 8; ++r) acc.f[r] = 0.f;
  const unsigned short* arow  = xch + (size_t)(t * 16) * DINN;
  const unsigned short* bbase = Wxp + (size_t)nb * 512 + lane * 16;

  FragB16 a0, a1, b0, b1;
  load_a(a0, arow, DINN, m, kh);
  load_b(b0, bbase);
  for (int kb = 0; kb < K32; kb += 2) {
    const int kb1 = kb + 1;
    const int kb2 = (kb + 2 < K32) ? kb + 2 : 0;
    load_a(a1, arow + kb1 * 32, DINN, m, kh);
    load_b(b1, bbase + (size_t)(kb1 * 6) * 512);
    acc.v = __builtin_amdgcn_wmma_f32_16x16x32_bf16(false, a0.v, false, b0.v,
                                                    (short)0, acc.v, false, false);
    load_a(a0, arow + kb2 * 32, DINN, m, kh);
    load_b(b0, bbase + (size_t)(kb2 * 6) * 512);
    acc.v = __builtin_amdgcn_wmma_f32_16x16x32_bf16(false, a1.v, false, b1.v,
                                                    (short)0, acc.v, false, false);
  }
  int col = nb * 16 + (lane & 15);
#pragma unroll
  for (int r = 0; r < 8; ++r)
    xp[(size_t)(t * 16 + (kh << 3) + r) * NXP + col] = acc.f[r];
}

// ---------------------------------------------------------------------------
// K5: dt = softplus(dt_in @ W_dt + b_dt)  (M=16384, K=64, N=2048).
// A gathered from f32 xp cols [0,64) with on-the-fly bf16 convert.
// ---------------------------------------------------------------------------
__global__ __launch_bounds__(256) void gemm_dt(const float* __restrict__ xp,
                                               const unsigned short* __restrict__ Wdtp,
                                               const float* __restrict__ b_dt,
                                               float* __restrict__ dt) {
  const int lane = threadIdx.x & 31;
  const int wid  = (blockIdx.x << 3) + (threadIdx.x >> 5);
  const int nb = wid >> 10;    // 16 groups of 128 cols (shared within block)
  const int t  = wid & 1023;
  const int m = lane & 15, kh = lane >> 4;
  AccF acc[8];
#pragma unroll
  for (int j = 0; j < 8; ++j)
#pragma unroll
    for (int r = 0; r < 8; ++r) acc[j].f[r] = 0.f;

  const float* arow = xp + (size_t)(t * 16) * NXP;
#pragma unroll
  for (int kb = 0; kb < 2; ++kb) {
    FragB16 a;
#pragma unroll
    for (int p = 0; p < 8; ++p) {
      int k = ((p & 3) << 1) + ((p >> 2) << 4) + (kh << 3) + kb * 32;
      a.h[2 * p]     = f2b(arow[(size_t)m * NXP + k]);
      a.h[2 * p + 1] = f2b(arow[(size_t)m * NXP + k + 1]);
    }
#pragma unroll
    for (int j = 0; j < 8; ++j) {
      FragB16 b;
      load_b(b, Wdtp + ((size_t)(kb * 128) + (nb * 8 + j)) * 512 + lane * 16);
      acc[j].v = __builtin_amdgcn_wmma_f32_16x16x32_bf16(false, a.v, false, b.v,
                                                         (short)0, acc[j].v, false, false);
    }
  }
#pragma unroll
  for (int j = 0; j < 8; ++j) {
    int col = nb * 128 + j * 16 + (lane & 15);
    float bias = b_dt[col];
#pragma unroll
    for (int r = 0; r < 8; ++r) {
      float v = acc[j].f[r] + bias;
      float sp = (v > 20.f) ? v : log1pf(__expf(v));
      dt[(size_t)(t * 16 + (kh << 3) + r) * DINN + col] = sp;
    }
  }
}

// ---------------------------------------------------------------------------
// K6: the only sequential part. 32768 lanes, one per (b,di); DS=16 states in
// registers; B/C vectors staged in LDS per step; writes u = y*silu(z) in bf16.
// ---------------------------------------------------------------------------
__global__ __launch_bounds__(256) void ssm_scan(const float* __restrict__ xp,
                                                const float* __restrict__ dt,
                                                const unsigned short* __restrict__ xch,
                                                const unsigned short* __restrict__ zh,
                                                const float* __restrict__ A_log,
                                                const float* __restrict__ Dv,
                                                const int* __restrict__ dones,
                                                unsigned short* __restrict__ uh) {
  __shared__ float sBC[32];    // [0,16)=Bm, [16,32)=Cm
  const int b  = blockIdx.x >> 3;
  const int di = ((blockIdx.x & 7) << 8) + threadIdx.x;

  float an[DSTT];
#pragma unroll
  for (int s = 0; s < DSTT; ++s) an[s] = -__expf(A_log[(size_t)di * DSTT + s]);
  const float dcoef = Dv[di];
  float st[DSTT];
#pragma unroll
  for (int s = 0; s < DSTT; ++s) st[s] = 0.f;

  for (int t = 0; t < S_LEN; ++t) {
    if (threadIdx.x < 32)
      sBC[threadIdx.x] = xp[(size_t)(t * BSZ + b) * NXP + DTRK + threadIdx.x];
    __syncthreads();
    float keep = 1.f;
    if (t > 0 && dones[(t - 1) * BSZ + b] != 0) keep = 0.f;
    size_t idx = (size_t)(t * BSZ + b) * DINN + di;
    float dtv = dt[idx];
    float xc  = b2f(xch[idx]);
    float zf  = b2f(zh[idx]);
    float y = dcoef * xc;
#pragma unroll
    for (int s = 0; s < DSTT; ++s) {
      float dA = __expf(dtv * an[s]);
      float sv = dA * (keep * st[s]) + dtv * sBC[s] * xc;
      st[s] = sv;
      y += sv * sBC[16 + s];
    }
    float sz = zf / (1.f + __expf(-zf));
    uh[idx] = f2b(y * sz);
    __syncthreads();
  }
}

// ---------------------------------------------------------------------------
// K7: out = u @ W_out  (M=16384, K=2048, N=1024). One wave -> 16x128 tile;
// LDS-staged B, 8 waves/block share one column group.
// ---------------------------------------------------------------------------
__global__ __launch_bounds__(256) void gemm_out(const unsigned short* __restrict__ uh,
                                                const unsigned short* __restrict__ Woutp,
                                                float* __restrict__ out) {
  __shared__ unsigned short sB[2 * 8 * 512];   // 16 KB double buffer
  const int lane = threadIdx.x & 31;
  const int wv   = threadIdx.x >> 5;
  const int wid  = (blockIdx.x << 3) + wv;
  const int nb = wid >> 10;    // 8 groups of 128 cols (uniform within block)
  const int t  = wid & 1023;
  const int m = lane & 15, kh = lane >> 4;

  AccF acc[8];
#pragma unroll
  for (int j = 0; j < 8; ++j)
#pragma unroll
    for (int r = 0; r < 8; ++r) acc[j].f[r] = 0.f;

  const unsigned short* arow   = uh + (size_t)(t * 16) * DINN;
  const unsigned short* btiles = Woutp + (size_t)(nb * 8) * 512;
  gemm_core_lds<64>(arow, DINN, btiles, sB, acc, DINN / 32, m, kh, lane, wv);

#pragma unroll
  for (int j = 0; j < 8; ++j) {
    int col = nb * 128 + j * 16 + (lane & 15);
#pragma unroll
    for (int r = 0; r < 8; ++r)
      out[(size_t)(t * 16 + (kh << 3) + r) * DMDL + col] = acc[j].f[r];
  }
}

// ---------------------------------------------------------------------------
extern "C" void kernel_launch(void* const* d_in, const int* in_sizes, int n_in,
                              void* d_out, int out_size, void* d_ws, size_t ws_size,
                              hipStream_t stream) {
  (void)in_sizes; (void)n_in; (void)out_size; (void)ws_size;
  const float* x_seq  = (const float*)d_in[0];
  const float* W_in   = (const float*)d_in[1];
  const float* conv_w = (const float*)d_in[2];
  const float* conv_b = (const float*)d_in[3];
  const float* W_x    = (const float*)d_in[4];
  const float* W_dt   = (const float*)d_in[5];
  const float* b_dt   = (const float*)d_in[6];
  const float* A_log  = (const float*)d_in[7];
  const float* Dvec   = (const float*)d_in[8];
  const float* W_out  = (const float*)d_in[9];
  const int*   dones  = (const int*)d_in[10];
  float* out = (float*)d_out;

  size_t off = 0;
  auto carve = [&](size_t bytes) -> void* {
    void* p = (char*)d_ws + off;
    off += (bytes + 255) & ~(size_t)255;
    return p;
  };
  unsigned short* xh    = (unsigned short*)carve((size_t)S_LEN * BSZ * DMDL * 2);
  unsigned short* Winp  = (unsigned short*)carve((size_t)DMDL * 2 * DINN * 2);
  unsigned short* Wxp   = (unsigned short*)carve((size_t)DINN * NXP * 2);
  unsigned short* Wdtp  = (unsigned short*)carve((size_t)DTRK * DINN * 2);
  unsigned short* Woutp = (unsigned short*)carve((size_t)DINN * DMDL * 2);
  unsigned short* xpath = (unsigned short*)carve((size_t)S_LEN * BSZ * DINN * 2);
  unsigned short* zh    = (unsigned short*)carve((size_t)S_LEN * BSZ * DINN * 2);
  unsigned short* xch   = (unsigned short*)carve((size_t)S_LEN * BSZ * DINN * 2);
  float*          xpw   = (float*)carve((size_t)S_LEN * BSZ * NXP * 4);
  float*          dtw   = (float*)carve((size_t)S_LEN * BSZ * DINN * 4);
  unsigned short* uhw   = (unsigned short*)carve((size_t)S_LEN * BSZ * DINN * 2);

  pack_w<<<4096, 256, 0, stream>>>(W_in,  Winp,  DMDL, 2 * DINN);
  pack_w<<<512,  256, 0, stream>>>(W_x,   Wxp,   DINN, NXP);
  pack_w<<<512,  256, 0, stream>>>(W_dt,  Wdtp,  DTRK, DINN);
  pack_w<<<2048, 256, 0, stream>>>(W_out, Woutp, DINN, DMDL);
  cvt_bf16<<<8192, 256, 0, stream>>>(x_seq, xh, (size_t)S_LEN * BSZ * DMDL);

  gemm_xz  <<<4096,  256, 0, stream>>>(xh, Winp, xpath, zh);
  conv_silu<<<16384, 256, 0, stream>>>(xpath, conv_w, conv_b, dones, xch);
  gemm_xp  <<<768,   256, 0, stream>>>(xch, Wxp, xpw);
  gemm_dt  <<<2048,  256, 0, stream>>>(xpw, Wdtp, b_dt, dtw);
  ssm_scan <<<128,   256, 0, stream>>>(xpw, dtw, xch, zh, A_log, Dvec, dones, uhw);
  gemm_out <<<1024,  256, 0, stream>>>(uhw, Woutp, out);
}
